// TAHGConv_27281632264463
// MI455X (gfx1250) — matrix operand order repaired and checked by reading, not compiled, not run on hardware
//
#include <hip/hip_runtime.h>
#include <hip/hip_bf16.h>

#define Nn   8
#define Vv   4096
#define Ee   2048
#define CIN  256
#define COUT 256

typedef __bf16 bf16_t;
typedef __attribute__((ext_vector_type(4)))  __bf16 bf16x4;
typedef __attribute__((ext_vector_type(8)))  __bf16 bf16x8;
typedef __attribute__((ext_vector_type(16))) __bf16 v16bf;
typedef __attribute__((ext_vector_type(8)))  float  v8f;
typedef __attribute__((ext_vector_type(4)))  unsigned int u32x4;
typedef __attribute__((ext_vector_type(4)))  int  i32x4;
typedef __attribute__((ext_vector_type(8)))  int  i32x8;

#define LDK 48   // LDS row stride in bf16 (64B data + 32B pad, inserted by the TDM)

// ---------------------------------------------------------------------------
// TDM: async 2D tile load global->LDS (CDNA5 ISA/§8).
// Tile = rows x rowlen bf16, source row stride = stride_elems.
// LDS gets 64B rows padded by 32B: pad_interval=3 (16 DWORDs), pad_amount=7
// (8 DWORDs) -> effective LDS row stride = LDK bf16. Tracked by TENSORcnt.
// ---------------------------------------------------------------------------
__device__ __forceinline__ void tdm_load_2d(unsigned int lds_byte_addr, const bf16_t* gptr,
                                            unsigned int rows, unsigned int rowlen,
                                            unsigned int stride_elems,
                                            unsigned int tensor_d0, unsigned int tensor_d1) {
    const unsigned long long ga = (unsigned long long)(size_t)gptr;
    u32x4 g0;
    g0[0] = 1u;                                                    // count=1 (valid user D#)
    g0[1] = lds_byte_addr;                                         // lds_addr
    g0[2] = (unsigned int)ga;                                      // global_addr[31:0]
    g0[3] = (unsigned int)((ga >> 32) & 0x1FFFFFFu) | (2u << 30);  // addr[56:32] | type=2
    i32x8 g1;
    g1[0] = (int)((1u << 16) | (1u << 20) | (3u << 22) | (7u << 25)); // data_size=2B, pad 16DW->+8DW
    g1[1] = (int)((tensor_d0 & 0xFFFFu) << 16);                    // tensor_dim0[15:0] @ bits63:48
    g1[2] = (int)((tensor_d0 >> 16) | ((tensor_d1 & 0xFFFFu) << 16)); // dim0 hi | dim1 lo
    g1[3] = (int)((tensor_d1 >> 16) | (rowlen << 16));             // dim1 hi | tile_dim0
    g1[4] = (int)(rows & 0xFFFFu);                                 // tile_dim1 | tile_dim2=0
    g1[5] = (int)stride_elems;                                     // tensor_dim0_stride[31:0]
    g1[6] = 0;                                                     // stride0 hi | stride1 lo
    g1[7] = 0;
    const i32x4 gz = {0, 0, 0, 0};
#if defined(__clang_major__) && (__clang_major__ >= 23)
    const i32x8 gz8 = {0, 0, 0, 0, 0, 0, 0, 0};
    __builtin_amdgcn_tensor_load_to_lds(g0, g1, gz, gz, gz8, 0);
#else
    __builtin_amdgcn_tensor_load_to_lds(g0, g1, gz, gz, 0);
#endif
}

// ---------------------------------------------------------------------------
// WMMA fragments out of LDS tiles stored K-contiguous per outer index.
// A fragment: lane r(=row M), two contiguous 16B chunks (K={h*8..+7,16+h*8..+7})
// B fragment: lane r(=col N), two contiguous 16B chunks (K={h*8..+7,16+h*8..+7})
// Both lower to 2x ds_load_b128.  C/D: VGPR r -> M=r+8*half, N=lane&15.
// ---------------------------------------------------------------------------
__device__ __forceinline__ v16bf frag_a(const bf16_t* lds, int outer, int lane) {
    const int r = lane & 15, half = lane >> 4;
    const bf16_t* row = lds + (outer + r) * LDK;
    v16bf f;
#pragma unroll
    for (int j = 0; j < 8; ++j) {
        f[j]     = row[half * 8 + j];
        f[j + 8] = row[16 + half * 8 + j];
    }
    return f;
}

__device__ __forceinline__ v16bf frag_b(const bf16_t* lds, int outer, int lane) {
    const int r = lane & 15, half = lane >> 4;
    const bf16_t* row = lds + (outer + r) * LDK;
    const bf16x8 lo = *(const bf16x8*)(row + half * 8);
    const bf16x8 hi = *(const bf16x8*)(row + 16 + half * 8);
    return __builtin_shufflevector(lo, hi, 0, 1, 2, 3, 4, 5, 6, 7, 8, 9, 10, 11, 12, 13, 14, 15);
}

#define WMMA_BF16(A, B, C) \
    __builtin_amdgcn_wmma_f32_16x16x32_bf16(false, (A), false, (B), (short)0, (C), false, false)

__device__ __forceinline__ float gelu_exact(float v) {
    return 0.5f * v * (1.0f + erff(v * 0.70710678118654752f));
}

// ---------------------------------------------------------------------------
// Pass 1: read fp32 H once; emit bf16 Hb and rowsum_inv[n,v] = 1/sum_e(H*ge)
// ---------------------------------------------------------------------------
__global__ __launch_bounds__(256)
void row_pass_kernel(const float* __restrict__ H, const float* __restrict__ ge,
                     bf16_t* __restrict__ Hb, float* __restrict__ rs_inv) {
    const int n = blockIdx.y, v = blockIdx.x;
    const size_t base = ((size_t)n * Vv + v) * Ee;
    const float* gen = ge + (size_t)n * Ee;
    float s = 0.f;
    for (int e4 = threadIdx.x * 4; e4 < Ee; e4 += 256 * 4) {
        const float4 h = *(const float4*)(H + base + e4);
        const float4 g = *(const float4*)(gen + e4);
        s += h.x * g.x + h.y * g.y + h.z * g.z + h.w * g.w;
        bf16x4 hb;
        hb[0] = (bf16_t)h.x; hb[1] = (bf16_t)h.y; hb[2] = (bf16_t)h.z; hb[3] = (bf16_t)h.w;
        *(bf16x4*)(Hb + base + e4) = hb;
    }
    __shared__ float red[256];
    red[threadIdx.x] = s;
    __syncthreads();
    for (int off = 128; off > 0; off >>= 1) {
        if (threadIdx.x < off) red[threadIdx.x] += red[threadIdx.x + off];
        __syncthreads();
    }
    if (threadIdx.x == 0) rs_inv[(size_t)n * Vv + v] = 1.0f / red[0];
}

// ---------------------------------------------------------------------------
// Pass 2: HbT[n,e,v] = Hb[n,v,e]  (tiled 64x64 transpose through LDS)
// ---------------------------------------------------------------------------
__global__ __launch_bounds__(256)
void transpose_kernel(const bf16_t* __restrict__ Hb, bf16_t* __restrict__ HbT) {
    const int n = blockIdx.z;
    const int vb = blockIdx.y * 64, eb = blockIdx.x * 64;
    __shared__ bf16_t tile[64][64 + 8];
    const bf16_t* src = Hb + (size_t)n * Vv * Ee;
    bf16_t* dst = HbT + (size_t)n * Vv * Ee;
    for (int idx = threadIdx.x; idx < 64 * 16; idx += 256) {
        const int row = idx >> 4, c4 = (idx & 15) << 2;
        const bf16x4 v = *(const bf16x4*)(src + (size_t)(vb + row) * Ee + eb + c4);
        tile[row][c4] = v[0]; tile[row][c4 + 1] = v[1];
        tile[row][c4 + 2] = v[2]; tile[row][c4 + 3] = v[3];
    }
    __syncthreads();
    for (int idx = threadIdx.x; idx < 64 * 16; idx += 256) {
        const int row = idx >> 4, c4 = (idx & 15) << 2;   // row = e, c4 = v chunk
        bf16x4 o;
        o[0] = tile[c4][row]; o[1] = tile[c4 + 1][row];
        o[2] = tile[c4 + 2][row]; o[3] = tile[c4 + 3][row];
        *(bf16x4*)(dst + (size_t)(eb + row) * Vv + vb + c4) = o;
    }
}

// ---------------------------------------------------------------------------
// Pass 3: colsum_inv[n,e] = 1/sum_v(HbT[e,v]*ns[v])   (coalesced row reduce)
// ---------------------------------------------------------------------------
__global__ __launch_bounds__(256)
void col_pass_kernel(const bf16_t* __restrict__ HbT, const float* __restrict__ ns,
                     float* __restrict__ cs_inv) {
    const int n = blockIdx.y, e = blockIdx.x;
    const bf16_t* row = HbT + ((size_t)n * Ee + e) * Vv;
    const float* nsn = ns + (size_t)n * Vv;
    float s = 0.f;
    for (int v4 = threadIdx.x * 4; v4 < Vv; v4 += 256 * 4) {
        const bf16x4 h = *(const bf16x4*)(row + v4);
        const float4 w = *(const float4*)(nsn + v4);
        s += (float)h[0] * w.x + (float)h[1] * w.y + (float)h[2] * w.z + (float)h[3] * w.w;
    }
    __shared__ float red[256];
    red[threadIdx.x] = s;
    __syncthreads();
    for (int off = 128; off > 0; off >>= 1) {
        if (threadIdx.x < off) red[threadIdx.x] += red[threadIdx.x + off];
        __syncthreads();
    }
    if (threadIdx.x == 0) cs_inv[(size_t)n * Ee + e] = 1.0f / red[0];
}

// ---------------------------------------------------------------------------
// Pass 4: thetaT[c,k] = bf16(theta[k,c]);  Pass 5: xb = bf16(x)
// ---------------------------------------------------------------------------
__global__ __launch_bounds__(256)
void theta_t_kernel(const float* __restrict__ theta, bf16_t* __restrict__ thT) {
    const int kb = blockIdx.y * 64, cb = blockIdx.x * 64;
    __shared__ bf16_t tile[64][64 + 8];
    for (int idx = threadIdx.x; idx < 64 * 16; idx += 256) {
        const int row = idx >> 4, c4 = (idx & 15) << 2;
        const float4 v = *(const float4*)(theta + (size_t)(kb + row) * COUT + cb + c4);
        tile[row][c4] = (bf16_t)v.x; tile[row][c4 + 1] = (bf16_t)v.y;
        tile[row][c4 + 2] = (bf16_t)v.z; tile[row][c4 + 3] = (bf16_t)v.w;
    }
    __syncthreads();
    for (int idx = threadIdx.x; idx < 64 * 16; idx += 256) {
        const int row = idx >> 4, c4 = (idx & 15) << 2;   // row = c, c4 = k chunk
        bf16x4 o;
        o[0] = tile[c4][row]; o[1] = tile[c4 + 1][row];
        o[2] = tile[c4 + 2][row]; o[3] = tile[c4 + 3][row];
        *(bf16x4*)(thT + (size_t)(cb + row) * CIN + kb + c4) = o;
    }
}

__global__ __launch_bounds__(256)
void xb_kernel(const float* __restrict__ x, bf16_t* __restrict__ xb) {
    const size_t i4 = ((size_t)blockIdx.x * 256 + threadIdx.x) * 4;
    const float4 v = *(const float4*)(x + i4);
    bf16x4 o;
    o[0] = (bf16_t)v.x; o[1] = (bf16_t)v.y; o[2] = (bf16_t)v.z; o[3] = (bf16_t)v.w;
    *(bf16x4*)(xb + i4) = o;
}

// ---------------------------------------------------------------------------
// Shared GEMM skeleton: macro tile 128x128, BK=32, 8 waves (wave tile 32x64),
// double-buffered TDM staging issued by wave 0.
// ---------------------------------------------------------------------------
#define GEMM_PROLOGUE(Aptr, Bptr, Kdim)                                              \
    const int tid = threadIdx.x;                                                     \
    const int lane = tid & 31, wave = tid >> 5;                                      \
    const int wm = (wave & 3) * 32, wn = (wave >> 2) * 64;                           \
    __shared__ bf16_t ldsA[2][128 * LDK];                                            \
    __shared__ bf16_t ldsB[2][128 * LDK];                                            \
    v8f acc[2][4] = {};                                                              \
    int cur = 0;                                                                     \
    if (wave == 0) {                                                                 \
        tdm_load_2d((unsigned int)(size_t)&ldsA[0][0], (Aptr) + (size_t)mblock * (Kdim), \
                    128, 32, (Kdim), (Kdim), 128);                                   \
        tdm_load_2d((unsigned int)(size_t)&ldsB[0][0], (Bptr) + (size_t)nblock * (Kdim), \
                    128, 32, (Kdim), (Kdim), 128);                                   \
    }                                                                                \
    for (int kb = 0; kb < (Kdim); kb += 32, cur ^= 1) {                              \
        if (wave == 0) {                                                             \
            if (kb + 32 < (Kdim)) {                                                  \
                tdm_load_2d((unsigned int)(size_t)&ldsA[cur ^ 1][0],                 \
                            (Aptr) + (size_t)mblock * (Kdim) + kb + 32,              \
                            128, 32, (Kdim), (Kdim), 128);                           \
                tdm_load_2d((unsigned int)(size_t)&ldsB[cur ^ 1][0],                 \
                            (Bptr) + (size_t)nblock * (Kdim) + kb + 32,              \
                            128, 32, (Kdim), (Kdim), 128);                           \
                __builtin_amdgcn_s_wait_tensorcnt((short)2);                         \
            } else {                                                                 \
                __builtin_amdgcn_s_wait_tensorcnt((short)0);                         \
            }                                                                        \
        }                                                                            \
        __syncthreads();                                                             \
        const v16bf a0 = frag_a(ldsA[cur], wm, lane);                                \
        const v16bf a1 = frag_a(ldsA[cur], wm + 16, lane);                           \
        _Pragma("unroll")                                                            \
        for (int ni = 0; ni < 4; ++ni) {                                             \
            const v16bf b = frag_b(ldsB[cur], wn + ni * 16, lane);                   \
            acc[0][ni] = WMMA_BF16(a0, b, acc[0][ni]);                               \
            acc[1][ni] = WMMA_BF16(a1, b, acc[1][ni]);                               \
        }                                                                            \
        __syncthreads();                                                             \
    }                                                                                \
    const int r = lane & 15, half = lane >> 4;

// GEMM 1: ybT[n,c,v] = bf16( ns[n,v] * (x[n,v,:] @ theta[:,c]) )
__global__ __launch_bounds__(256)
void gemm_y_kernel(const bf16_t* __restrict__ xb, const bf16_t* __restrict__ thT,
                   const float* __restrict__ ns, bf16_t* __restrict__ ybT) {
    const int n = blockIdx.z;
    const int mblock = blockIdx.y * 128;   // v
    const int nblock = blockIdx.x * 128;   // c
    const bf16_t* An = xb + (size_t)n * Vv * CIN;
    GEMM_PROLOGUE(An, thT, CIN)
    const float* nsn = ns + (size_t)n * Vv;
#pragma unroll
    for (int mi = 0; mi < 2; ++mi)
#pragma unroll
        for (int ni = 0; ni < 4; ++ni) {
            const int c = nblock + wn + ni * 16 + r;
            const int m0 = mblock + wm + mi * 16 + half * 8;
            bf16x8 ob;
#pragma unroll
            for (int e = 0; e < 8; ++e)
                ob[e] = (bf16_t)(acc[mi][ni][e] * nsn[m0 + e]);
            *(bf16x8*)(ybT + ((size_t)n * COUT + c) * Vv + m0) = ob;
        }
}

// GEMM 2: e_fts[n,e,c] = cs_inv*(HbT @ ybT^T);  ebT[n,c,e] = bf16(ge*e_fts)
__global__ __launch_bounds__(256)
void gemm_efts_kernel(const bf16_t* __restrict__ HbT, const bf16_t* __restrict__ ybT,
                      const float* __restrict__ cs_inv, const float* __restrict__ ge,
                      float* __restrict__ efts, bf16_t* __restrict__ ebT) {
    const int n = blockIdx.z;
    const int mblock = blockIdx.y * 128;   // e
    const int nblock = blockIdx.x * 128;   // c
    const bf16_t* An = HbT + (size_t)n * Ee * Vv;
    const bf16_t* Bn = ybT + (size_t)n * COUT * Vv;
    GEMM_PROLOGUE(An, Bn, Vv)
#pragma unroll
    for (int mi = 0; mi < 2; ++mi)
#pragma unroll
        for (int ni = 0; ni < 4; ++ni) {
            const int c = nblock + wn + ni * 16 + r;
            const int m0 = mblock + wm + mi * 16 + half * 8;   // hyperedge base
            bf16x8 ob;
#pragma unroll
            for (int e = 0; e < 8; ++e) {
                const int m = m0 + e;
                const float val = acc[mi][ni][e] * cs_inv[(size_t)n * Ee + m];
                efts[((size_t)n * Ee + m) * COUT + c] = val;
                ob[e] = (bf16_t)(val * ge[(size_t)n * Ee + m]);
            }
            *(bf16x8*)(ebT + ((size_t)n * COUT + c) * Ee + m0) = ob;
        }
}

// GEMM 3: out[n,v,c] = gelu( rs_inv*(Hb @ ebT^T) + bias )
__global__ __launch_bounds__(256)
void gemm_y2_kernel(const bf16_t* __restrict__ Hb, const bf16_t* __restrict__ ebT,
                    const float* __restrict__ rs_inv, const float* __restrict__ bias,
                    float* __restrict__ out) {
    const int n = blockIdx.z;
    const int mblock = blockIdx.y * 128;   // v
    const int nblock = blockIdx.x * 128;   // c
    const bf16_t* An = Hb + (size_t)n * Vv * Ee;
    const bf16_t* Bn = ebT + (size_t)n * COUT * Ee;
    GEMM_PROLOGUE(An, Bn, Ee)
#pragma unroll
    for (int mi = 0; mi < 2; ++mi)
#pragma unroll
        for (int ni = 0; ni < 4; ++ni) {
            const int c = nblock + wn + ni * 16 + r;
            const float bc = bias[c];
            const int m0 = mblock + wm + mi * 16 + half * 8;   // node base
#pragma unroll
            for (int e = 0; e < 8; ++e) {
                const int m = m0 + e;
                const float val = acc[mi][ni][e] * rs_inv[(size_t)n * Vv + m] + bc;
                out[((size_t)n * Vv + m) * COUT + c] = gelu_exact(val);
            }
        }
}

// ---------------------------------------------------------------------------
extern "C" void kernel_launch(void* const* d_in, const int* in_sizes, int n_in,
                              void* d_out, int out_size, void* d_ws, size_t ws_size,
                              hipStream_t stream) {
    const float* x     = (const float*)d_in[0];   // [N,V,C_IN]
    const float* H     = (const float*)d_in[1];   // [N,V,E]
    const float* ns    = (const float*)d_in[2];   // [N,V,1]
    const float* ge    = (const float*)d_in[3];   // [N,E]
    const float* theta = (const float*)d_in[4];   // [C_IN,C_OUT]
    const float* bias  = (const float*)d_in[5];   // [C_OUT]

    float* out_y2   = (float*)d_out;                         // [N,V,C_OUT]
    float* out_efts = out_y2 + (size_t)Nn * Vv * COUT;       // [N,E,C_OUT]

    char* ws = (char*)d_ws;
    size_t off = 0;
    bf16_t* Hb  = (bf16_t*)(ws + off); off += (size_t)Nn * Vv * Ee   * sizeof(bf16_t); // 128 MB
    bf16_t* HbT = (bf16_t*)(ws + off); off += (size_t)Nn * Vv * Ee   * sizeof(bf16_t); // 128 MB
    bf16_t* xbw = (bf16_t*)(ws + off); off += (size_t)Nn * Vv * CIN  * sizeof(bf16_t); //  16 MB
    bf16_t* ybT = (bf16_t*)(ws + off); off += (size_t)Nn * Vv * COUT * sizeof(bf16_t); //  16 MB
    bf16_t* ebT = (bf16_t*)(ws + off); off += (size_t)Nn * Ee * COUT * sizeof(bf16_t); //   8 MB
    bf16_t* thT = (bf16_t*)(ws + off); off += (size_t)CIN * COUT     * sizeof(bf16_t);
    float* rs_inv = (float*)(ws + off); off += (size_t)Nn * Vv * sizeof(float);
    float* cs_inv = (float*)(ws + off); off += (size_t)Nn * Ee * sizeof(float);
    (void)ws_size; (void)in_sizes; (void)n_in; (void)out_size;

    row_pass_kernel<<<dim3(Vv, Nn), 256, 0, stream>>>(H, ge, Hb, rs_inv);
    transpose_kernel<<<dim3(Ee / 64, Vv / 64, Nn), 256, 0, stream>>>(Hb, HbT);
    col_pass_kernel<<<dim3(Ee, Nn), 256, 0, stream>>>(HbT, ns, cs_inv);
    theta_t_kernel<<<dim3(COUT / 64, CIN / 64), 256, 0, stream>>>(theta, thT);
    xb_kernel<<<dim3((Nn * Vv * CIN) / (256 * 4)), 256, 0, stream>>>(x, xbw);
    gemm_y_kernel<<<dim3(COUT / 128, Vv / 128, Nn), 256, 0, stream>>>(xbw, thT, ns, ybT);
    gemm_efts_kernel<<<dim3(COUT / 128, Ee / 128, Nn), 256, 0, stream>>>(HbT, ybT, cs_inv, ge, out_efts, ebT);
    gemm_y2_kernel<<<dim3(COUT / 128, Vv / 128, Nn), 256, 0, stream>>>(Hb, ebT, rs_inv, bias, out_y2);
}
